// ComplementaryChannelInteraction_13460427506002
// MI455X (gfx1250) — matrix operand order repaired and checked by reading, not compiled, not run on hardware
//
#include <hip/hip_runtime.h>
#include <hip/hip_bf16.h>
#include <stdint.h>

// ---------------------------------------------------------------------------
// CDNA5 (gfx1250) bf16 WMMA implementation of the ComplementaryChannel block.
// All GEMMs use V_WMMA_F32_16X16X32_BF16 (wave32, f32 accumulate).
// Flash kernel stages its LDS tile with the Tensor Data Mover (TDM).
// ---------------------------------------------------------------------------

typedef __attribute__((ext_vector_type(16))) __bf16 v16bf;
typedef __attribute__((ext_vector_type(8)))  float  v8f;
typedef __attribute__((ext_vector_type(4)))  unsigned int v4u;
typedef __attribute__((ext_vector_type(8)))  int    v8i;
typedef __attribute__((ext_vector_type(4)))  int    v4i;

union BF16x16 { v16bf v; uint4 q[2]; };

static __device__ __forceinline__ v8f wmma_bf16(v16bf a, v16bf b, v8f c) {
  return __builtin_amdgcn_wmma_f32_16x16x32_bf16(false, a, false, b, (short)0, c,
                                                 false, false);
}

static __device__ __forceinline__ v8f vzero8() {
  v8f z = {0.f,0.f,0.f,0.f,0.f,0.f,0.f,0.f};
  return z;
}

// A-matrix 16x32 bf16 fragment from a row-major [16][stride] tile.
// K order used here (A and B match): lanes 0-15 -> M=lane, K = 0..15 ; lanes 16-31 -> K = 16..31.
static __device__ __forceinline__ v16bf load_fragA(const __bf16* base, int stride) {
  const int lane = threadIdx.x & 31;
  const int m  = lane & 15;
  const int kb = (lane & 16) ? 16 : 0;
  const __bf16* p = base + (size_t)m * stride + kb;
  BF16x16 f;
  f.q[0] = *(const uint4*)(p);        // K = kb .. kb+7
  f.q[1] = *(const uint4*)(p + 8);    // K = kb+8 .. kb+15
  return f.v;
}

// B-matrix 32x16 bf16 fragment from a [N][K] (row = output column n) tile.
// K order used here (A and B match): lanes 0-15 -> N=lane, K=0..15 ; lanes 16-31 -> K=16..31.
static __device__ __forceinline__ v16bf load_fragB(const __bf16* base, int stride) {
  const int lane = threadIdx.x & 31;
  const int n  = lane & 15;
  const int kb = (lane & 16) ? 16 : 0;
  const __bf16* p = base + (size_t)n * stride + kb;
  BF16x16 f;
  f.q[0] = *(const uint4*)(p);
  f.q[1] = *(const uint4*)(p + 8);
  return f.v;
}

// B fragment where each lane supplies its own row pointer (conv3 spatial taps).
static __device__ __forceinline__ v16bf load_fragB_row(const __bf16* rowbase) {
  const int lane = threadIdx.x & 31;
  const int kb = (lane & 16) ? 16 : 0;
  const __bf16* p = rowbase + kb;
  BF16x16 f;
  f.q[0] = *(const uint4*)(p);
  f.q[1] = *(const uint4*)(p + 8);
  return f.v;
}

// ---------------------------------------------------------------------------
// Precision-conversion / layout kernels
// ---------------------------------------------------------------------------

__global__ void __launch_bounds__(256) convert_wr(const float* __restrict__ w,
                                                  __bf16* __restrict__ o) {
  int i = blockIdx.x * 256 + threadIdx.x;
  if (i < 512 * 2048) o[i] = (__bf16)w[i];
}

// feature [128][2048][196] f32 -> featT [128][208][2048] bf16 (n-major, zero pad)
__global__ void __launch_bounds__(256) convert_feat(const float* __restrict__ f,
                                                    __bf16* __restrict__ o) {
  size_t i = (size_t)blockIdx.x * 256 + threadIdx.x;
  if (i >= (size_t)128 * 208 * 2048) return;
  int ci = (int)(i & 2047);
  int n  = (int)((i >> 11) % 208);
  int b  = (int)(i / ((size_t)208 * 2048));
  float v = (n < 196) ? f[((size_t)b * 2048 + ci) * 196 + n] : 0.0f;
  o[i] = (__bf16)v;
}

// w3 [512][512][3][3] f32 -> w3r [512][9][512] bf16  (K ordered (kh,kw,ci))
__global__ void __launch_bounds__(256) convert_w3(const float* __restrict__ w,
                                                  __bf16* __restrict__ o) {
  int i = blockIdx.x * 256 + threadIdx.x;
  if (i >= 512 * 4608) return;
  int co  = i / 4608;
  int rem = i % 4608;
  int pos = rem / 512;          // kh*3+kw
  int ci  = rem % 512;
  o[i] = (__bf16)w[((size_t)co * 512 + ci) * 9 + pos];
}

// ---------------------------------------------------------------------------
// Kernel 1: 1x1 reduce conv as GEMM  x[b] = ReLU(BN1(Wr @ feature[b] + b))
//   M=512 (co), K=2048 (ci), N=196 (spatial).  4 WGs/batch, 1 m-tile/wave.
// ---------------------------------------------------------------------------
__global__ void __launch_bounds__(256) gemm1_kernel(
    const __bf16* __restrict__ Wr, const __bf16* __restrict__ featT,
    const float* __restrict__ b_reduce,
    const float* __restrict__ g1, const float* __restrict__ be1,
    const float* __restrict__ mu1, const float* __restrict__ va1,
    __bf16* __restrict__ x_g)
{
  const int wg   = blockIdx.x;        // 512 = 128 batches * 4 row-blocks
  const int b    = wg >> 2;
  const int mblk = wg & 3;
  const int wave = threadIdx.x >> 5;
  const int lane = threadIdx.x & 31;
  const int c0   = mblk * 128 + wave * 16;

  const __bf16* Bbase = featT + (size_t)b * 208 * 2048;
  const __bf16* Abase = Wr + (size_t)c0 * 2048;

  v8f acc[13];
  #pragma unroll
  for (int t = 0; t < 13; ++t) acc[t] = vzero8();

  for (int kk = 0; kk < 64; ++kk) {
    __builtin_prefetch(Abase + (kk + 2) * 32, 0, 1);    // global_prefetch_b8
    v16bf a = load_fragA(Abase + kk * 32, 2048);
    #pragma unroll
    for (int t = 0; t < 13; ++t) {
      v16bf bb = load_fragB(Bbase + (size_t)(t * 16) * 2048 + kk * 32, 2048);
      acc[t] = wmma_bf16(a, bb, acc[t]);
    }
  }

  // Fused bias + BN1 + ReLU epilogue. D layout: VGPR r -> row (r | 8+r), lane -> n.
  const int half = (lane & 16) ? 8 : 0;
  float s[8], sh[8], br[8];
  #pragma unroll
  for (int r = 0; r < 8; ++r) {
    int c = c0 + half + r;
    float sv = g1[c] * rsqrtf(va1[c] + 1e-5f);
    s[r] = sv; sh[r] = be1[c] - mu1[c] * sv; br[r] = b_reduce[c];
  }
  #pragma unroll
  for (int t = 0; t < 13; ++t) {
    int n = t * 16 + (lane & 15);
    if (n < 196) {
      #pragma unroll
      for (int r = 0; r < 8; ++r) {
        int c = c0 + half + r;
        float v = fmaxf((acc[t][r] + br[r]) * s[r] + sh[r], 0.0f);
        x_g[((size_t)b * 512 + c) * 196 + n] = (__bf16)v;
      }
    }
  }
}

// ---------------------------------------------------------------------------
// Kernel 2: fused  xxt -> softmax(-xxt) -> a @ x   (flash-attention style)
//   x[b] (512x196 bf16) staged whole into LDS by the Tensor Data Mover:
//   one TENSOR_LOAD_TO_LDS with tensor=512x196, tile=512x224 -- the OOB
//   columns (n=196..223) are zero-filled by TDM, giving the K padding free.
//   Online softmax over 16 d-chunks of 32; y stored transposed [b][n][c] bf16.
// ---------------------------------------------------------------------------
__global__ void __launch_bounds__(256) flash_kernel(const __bf16* __restrict__ x_g,
                                                    __bf16* __restrict__ y_g)
{
  extern __shared__ char smem[];
  __bf16* x_lds  = (__bf16*)smem;                              // [512][224]
  __bf16* vt_lds = (__bf16*)(smem + (size_t)512 * 224 * 2);    // [224][32]
  __bf16* p_lds  = (__bf16*)(smem + (size_t)512 * 224 * 2 + 224 * 32 * 2); // [8][16][32]

  const int wg   = blockIdx.x;     // 512 = 128 batches * 4
  const int b    = wg >> 2;
  const int mblk = wg & 3;
  const int wave = threadIdx.x >> 5;
  const int lane = threadIdx.x & 31;
  const int tid  = threadIdx.x;

  const __bf16* xb = x_g + (size_t)b * 512 * 196;

  // --- TDM staging: one descriptor, issued by wave 0 ---------------------
  if (tid < 32) {
    uint64_t ga  = (uint64_t)(uintptr_t)xb;
    unsigned lds = (unsigned)(uintptr_t)x_lds;   // low 32 bits = LDS byte offset
    v4u g0;
    g0.x = 1u;                                        // count=1 (valid user D#)
    g0.y = lds;                                       // lds_addr
    g0.z = (unsigned)ga;                              // global_addr[31:0]
    g0.w = (unsigned)((ga >> 32) & 0x01FFFFFFu)       // global_addr[56:32]
         | (2u << 30);                                // type=2 ("image")
    v8i g1;
    g1[0] = 0x00010000;            // wg_mask=0, data_size=1 (2 bytes)
    g1[1] = (int)(196u << 16);     // tensor_dim0 = 196 (low 16 in [31:16])
    g1[2] = (int)(512u << 16);     // td0 hi=0 | tensor_dim1 = 512
    g1[3] = (int)(224u << 16);     // td1 hi=0 | tile_dim0 = 224
    g1[4] = 512;                   // tile_dim1 = 512, tile_dim2 = 0 (2D)
    g1[5] = 196;                   // tensor_dim0_stride[31:0]
    g1[6] = (int)0x88000000u;      // s0 hi=0 | tensor_dim1_stride[15:0]=0x8800
    g1[7] = 1;                     // tensor_dim1_stride[47:16] = 0x1 (100352)
    v4i gz  = {0, 0, 0, 0};
    v8i gz8 = {0, 0, 0, 0, 0, 0, 0, 0};
    __builtin_amdgcn_tensor_load_to_lds(g0, g1, gz, gz, gz8, 0);
    __builtin_amdgcn_s_wait_tensorcnt(0);
  }
  __syncthreads();

  const int c0 = mblk * 128 + wave * 16;
  float mrow[8], lrow[8];
  #pragma unroll
  for (int r = 0; r < 8; ++r) { mrow[r] = -3.0e38f; lrow[r] = 0.0f; }
  v8f acc[13];
  #pragma unroll
  for (int t = 0; t < 13; ++t) acc[t] = vzero8();
  __bf16* p_w = p_lds + wave * 16 * 32;

  for (int dc = 0; dc < 16; ++dc) {
    const int d0 = dc * 32;
    __syncthreads();                       // previous vt consumers done
    // Cooperative 224x32 transpose slice (x^T) -- packed 32-bit LDS stores.
    for (int i = tid; i < 224 * 16; i += 256) {
      int n = i >> 4, dd = (i & 15) * 2;
      unsigned short u0 = __builtin_bit_cast(unsigned short,
                            x_lds[(size_t)(d0 + dd) * 224 + n]);
      unsigned short u1 = __builtin_bit_cast(unsigned short,
                            x_lds[(size_t)(d0 + dd + 1) * 224 + n]);
      ((unsigned*)vt_lds)[(size_t)n * 16 + (dd >> 1)] =
          (unsigned)u0 | ((unsigned)u1 << 16);
    }
    __syncthreads();

    // S = x_m @ x_d^T over K=224 (7 chunks); both operands from x_lds row-major.
    v8f S0 = vzero8(), S1 = vzero8();
    #pragma unroll
    for (int kk = 0; kk < 7; ++kk) {
      v16bf a  = load_fragA(x_lds + (size_t)c0 * 224 + kk * 32, 224);
      v16bf b0 = load_fragB(x_lds + (size_t)d0 * 224 + kk * 32, 224);
      v16bf b1 = load_fragB(x_lds + (size_t)(d0 + 16) * 224 + kk * 32, 224);
      S0 = wmma_bf16(a, b0, S0);
      S1 = wmma_bf16(a, b1, S1);
    }

    // Online softmax over logits = -S  (row stats via half-wave shuffles).
    #pragma unroll
    for (int r = 0; r < 8; ++r) {
      float v0 = -S0[r], v1 = -S1[r];
      float mx = fmaxf(v0, v1);
      for (int m = 1; m < 16; m <<= 1) mx = fmaxf(mx, __shfl_xor(mx, m, 32));
      float mnew = fmaxf(mrow[r], mx);
      float corr = __expf(mrow[r] - mnew);
      float p0 = __expf(v0 - mnew);
      float p1 = __expf(v1 - mnew);
      float ps = p0 + p1;
      for (int m = 1; m < 16; m <<= 1) ps += __shfl_xor(ps, m, 32);
      lrow[r] = lrow[r] * corr + ps;
      mrow[r] = mnew;
      #pragma unroll
      for (int t = 0; t < 13; ++t) acc[t][r] *= corr;
      int prow = ((lane & 16) ? 8 : 0) + r;
      p_w[prow * 32 + (lane & 15)]      = (__bf16)p0;    // D-layout -> A-layout
      p_w[prow * 32 + 16 + (lane & 15)] = (__bf16)p1;
    }
    asm volatile("s_wait_dscnt 0" ::: "memory");         // cross-lane LDS RAW

    // acc += P(16x32) @ x_d(32x196): B from transposed slice, all-b128 loads.
    v16bf pa = load_fragA(p_w, 32);
    #pragma unroll
    for (int t = 0; t < 13; ++t) {
      v16bf bb = load_fragB(vt_lds + (size_t)(t * 16) * 32, 32);
      acc[t] = wmma_bf16(pa, bb, acc[t]);
    }
  }

  // Normalize and store y transposed [b][n][c] for conv3's im2col staging.
  const int half = (lane & 16) ? 8 : 0;
  __bf16* yb = y_g + (size_t)b * 196 * 512;
  #pragma unroll
  for (int t = 0; t < 13; ++t) {
    int n = t * 16 + (lane & 15);
    if (n < 196) {
      #pragma unroll
      for (int r = 0; r < 8; ++r) {
        float v = acc[t][r] / lrow[r];
        yb[(size_t)n * 512 + c0 + half + r] = (__bf16)v;
      }
    }
  }
}

// ---------------------------------------------------------------------------
// Kernel 3: 3x3 SAME conv as implicit GEMM + BN2 + ReLU + global avg pool.
//   yT staged in LDS with zero halo ring: [16x16 padded spatial][512 ci] bf16.
//   K ordered (kh,kw,ci) so B fragments are contiguous in ci.
// ---------------------------------------------------------------------------
__global__ void __launch_bounds__(256) conv3_kernel(
    const __bf16* __restrict__ y_g, const __bf16* __restrict__ w3r,
    const float* __restrict__ b3,
    const float* __restrict__ g2, const float* __restrict__ be2,
    const float* __restrict__ mu2, const float* __restrict__ va2,
    float* __restrict__ z)
{
  extern __shared__ char smem[];
  __bf16* yT = (__bf16*)smem;                 // [256][512]
  const int b    = blockIdx.x;
  const int tid  = threadIdx.x;
  const int wave = tid >> 5;
  const int lane = tid & 31;
  const __bf16* yb = y_g + (size_t)b * 196 * 512;

  for (int i = tid; i < 256 * 256; i += 256) {     // dword granularity fill
    int sp = i >> 8;
    int dw = i & 255;
    int hh = (sp >> 4) - 1, ww = (sp & 15) - 1;
    unsigned val = 0u;
    if (hh >= 0 && hh < 14 && ww >= 0 && ww < 14)
      val = ((const unsigned*)(yb + (size_t)(hh * 14 + ww) * 512))[dw];
    ((unsigned*)yT)[i] = val;
  }
  __syncthreads();

  int hn[13], wn[13];
  #pragma unroll
  for (int t = 0; t < 13; ++t) {
    int n = t * 16 + (lane & 15);
    if (n > 195) n = 195;                    // clamped; masked out of pooling
    hn[t] = n / 14; wn[t] = n % 14;
  }

  const int half = (lane & 16) ? 8 : 0;
  for (int mt = 0; mt < 4; ++mt) {           // 8 waves * 4 m-tiles = 512 co
    const int c0 = (wave * 4 + mt) * 16;
    v8f acc[13];
    #pragma unroll
    for (int t = 0; t < 13; ++t) acc[t] = vzero8();

    for (int kh = 0; kh < 3; ++kh)
      for (int kw = 0; kw < 3; ++kw) {
        const __bf16* abase = w3r + (size_t)c0 * 4608 + (kh * 3 + kw) * 512;
        for (int ck = 0; ck < 16; ++ck) {    // ci in chunks of 32
          v16bf a = load_fragA(abase + ck * 32, 4608);
          #pragma unroll
          for (int t = 0; t < 13; ++t) {
            int sp = (hn[t] + kh) * 16 + (wn[t] + kw);
            v16bf bb = load_fragB_row(yT + (size_t)sp * 512 + ck * 32);
            acc[t] = wmma_bf16(a, bb, acc[t]);
          }
        }
      }

    float s[8], sh[8], bc[8], rsum[8];
    #pragma unroll
    for (int r = 0; r < 8; ++r) {
      int c = c0 + half + r;
      float sv = g2[c] * rsqrtf(va2[c] + 1e-5f);
      s[r] = sv; sh[r] = be2[c] - mu2[c] * sv; bc[r] = b3[c]; rsum[r] = 0.f;
    }
    #pragma unroll
    for (int t = 0; t < 13; ++t) {
      int n = t * 16 + (lane & 15);
      float msk = (n < 196) ? 1.f : 0.f;
      #pragma unroll
      for (int r = 0; r < 8; ++r) {
        float v = fmaxf((acc[t][r] + bc[r]) * s[r] + sh[r], 0.f);
        rsum[r] += v * msk;
      }
    }
    #pragma unroll
    for (int r = 0; r < 8; ++r) {            // pool across lanes (spatial)
      float v = rsum[r];
      for (int m = 1; m < 16; m <<= 1) v += __shfl_xor(v, m, 32);
      if ((lane & 15) == 0)
        z[(size_t)b * 512 + c0 + half + r] = v * (1.0f / 196.0f);
    }
  }
}

// ---------------------------------------------------------------------------
// Kernel 4: FC -> ReLU -> FC  (tiny: 128x200x512 + 128x200x200, plain VALU)
// ---------------------------------------------------------------------------
__global__ void __launch_bounds__(256) fc_kernel(
    const float* __restrict__ z,
    const float* __restrict__ fc1w, const float* __restrict__ fc1b,
    const float* __restrict__ fc2w, const float* __restrict__ fc2b,
    float* __restrict__ out)
{
  __shared__ float zs[512];
  __shared__ float h[200];
  const int b = blockIdx.x, tid = threadIdx.x;
  for (int i = tid; i < 512; i += 256) zs[i] = z[(size_t)b * 512 + i];
  __syncthreads();
  for (int j = tid; j < 200; j += 256) {
    float a = fc1b[j];
    const float* w = fc1w + (size_t)j * 512;
    for (int k = 0; k < 512; ++k) a += zs[k] * w[k];
    h[j] = fmaxf(a, 0.f);
  }
  __syncthreads();
  for (int j = tid; j < 200; j += 256) {
    float a = fc2b[j];
    const float* w = fc2w + (size_t)j * 200;
    for (int k = 0; k < 200; ++k) a += h[k] * w[k];
    out[(size_t)b * 200 + j] = a;
  }
}

// ---------------------------------------------------------------------------
extern "C" void kernel_launch(void* const* d_in, const int* in_sizes, int n_in,
                              void* d_out, int out_size, void* d_ws, size_t ws_size,
                              hipStream_t stream) {
  const float* feature  = (const float*)d_in[0];
  const float* w_reduce = (const float*)d_in[1];
  const float* b_reduce = (const float*)d_in[2];
  const float* g1  = (const float*)d_in[3];
  const float* be1 = (const float*)d_in[4];
  const float* mu1 = (const float*)d_in[5];
  const float* va1 = (const float*)d_in[6];
  const float* w3  = (const float*)d_in[7];
  const float* b3  = (const float*)d_in[8];
  const float* g2  = (const float*)d_in[9];
  const float* be2 = (const float*)d_in[10];
  const float* mu2 = (const float*)d_in[11];
  const float* va2 = (const float*)d_in[12];
  const float* fc1w = (const float*)d_in[13];
  const float* fc1b = (const float*)d_in[14];
  const float* fc2w = (const float*)d_in[15];
  const float* fc2b = (const float*)d_in[16];
  float* out = (float*)d_out;

  // Workspace carve-out (~168 MB total).
  char* ws = (char*)d_ws;
  size_t off = 0;
  __bf16* Wr    = (__bf16*)(ws + off); off += (size_t)512 * 2048 * 2;        // 2 MB
  __bf16* featT = (__bf16*)(ws + off); off += (size_t)128 * 208 * 2048 * 2;  // 109 MB
  __bf16* x_g   = (__bf16*)(ws + off); off += (size_t)128 * 512 * 196 * 2;   // 25.7 MB
  __bf16* y_g   = (__bf16*)(ws + off); off += (size_t)128 * 196 * 512 * 2;   // 25.7 MB
  __bf16* w3r   = (__bf16*)(ws + off); off += (size_t)512 * 4608 * 2;        // 4.7 MB
  float*  zbuf  = (float*)(ws + off);  off += (size_t)128 * 512 * 4;         // 0.25 MB

  convert_wr  <<<(512 * 2048 + 255) / 256, 256, 0, stream>>>(w_reduce, Wr);
  convert_feat<<<(int)(((size_t)128 * 208 * 2048 + 255) / 256), 256, 0, stream>>>(feature, featT);
  convert_w3  <<<(512 * 4608 + 255) / 256, 256, 0, stream>>>(w3, w3r);

  gemm1_kernel<<<512, 256, 0, stream>>>(Wr, featT, b_reduce, g1, be1, mu1, va1, x_g);
  flash_kernel<<<512, 256, (size_t)512 * 224 * 2 + 224 * 32 * 2 + 8 * 16 * 32 * 2,
                stream>>>(x_g, y_g);
  conv3_kernel<<<128, 256, (size_t)256 * 512 * 2, stream>>>(y_g, w3r, b3, g2, be2,
                                                            mu2, va2, zbuf);
  fc_kernel<<<128, 256, 0, stream>>>(zbuf, fc1w, fc1b, fc2w, fc2b, out);
}
